// NeighborhoodAttentionModule_6923487282223
// MI455X (gfx1250) — matrix operand order repaired and checked by reading, not compiled, not run on hardware
//
#include <hip/hip_runtime.h>

// ---------------- types / helpers ----------------
typedef __attribute__((ext_vector_type(16))) __bf16 bf16x16;
typedef __attribute__((ext_vector_type(8)))  float  f32x8;
typedef __attribute__((ext_vector_type(4)))  float  f32x4;   // native vec for NT loads

__device__ __forceinline__ unsigned short f2bf(float f) {
  union { float f; unsigned u; } v; v.f = f;
  unsigned r = v.u + 0x7FFFu + ((v.u >> 16) & 1u);   // round-to-nearest-even
  return (unsigned short)(r >> 16);
}
__device__ __forceinline__ float bf2f(unsigned short b) {
  union { unsigned u; float f; } v; v.u = ((unsigned)b) << 16;
  return v.f;
}

union FragU { uint4 q[2]; bf16x16 v; };

// A fragment (16x32 bf16, M=row, K=col) from LDS, row-major with 'stride' (elems).
// Per ISA 7.12.2: lane<16 holds row m, K = k0+{0..7, 16..23}; lane>=16: K = k0+{8..15, 24..31}.
__device__ __forceinline__ bf16x16 load_a_lds(const unsigned short* base, int stride,
                                              int k0, int lane) {
  int m = lane & 15;
  int koff = k0 + ((lane & 16) ? 8 : 0);
  const unsigned short* p = base + m * stride + koff;
  FragU u;
  u.q[0] = *(const uint4*)(p);        // K koff..koff+7
  u.q[1] = *(const uint4*)(p + 16);   // K koff+16..koff+23
  return u.v;
}

// B fragment (32x16 bf16, K x N) where B[k][n] = W[row0+n][k] (nn.Linear x@W.T).
// Same K order as A: lanes 0-15 -> N=lane, K=k0+{0..7, 16..23};
// lanes 16-31 -> N=lane-16, K=k0+{8..15, 24..31}.
__device__ __forceinline__ bf16x16 load_b_glb(const unsigned short* base, int row0,
                                              int k0, int lane) {
  int n = lane & 15;
  int koff = k0 + ((lane & 16) ? 8 : 0);
  const unsigned short* p = base + (size_t)(row0 + n) * 256 + koff;
  FragU u;
  u.q[0] = *(const uint4*)(p);
  u.q[1] = *(const uint4*)(p + 16);
  return u.v;
}

__device__ __forceinline__ f32x8 wmma_bf16(bf16x16 a, bf16x16 b, f32x8 c) {
  return __builtin_amdgcn_wmma_f32_16x16x32_bf16(false, a, false, b, (short)0, c,
                                                 false, false);
}

// ---------------- prep kernels ----------------
// Convert Wv/Wc/Wx to bf16 in workspace.
__global__ __launch_bounds__(256) void na_prep_w(
    const float* __restrict__ Wv, const float* __restrict__ Wc,
    const float* __restrict__ Wx, unsigned short* __restrict__ oV,
    unsigned short* __restrict__ oC, unsigned short* __restrict__ oX) {
  int idx = blockIdx.x * 256 + threadIdx.x;      // 131072 total
  if (idx < 65536)       oV[idx]          = f2bf(Wv[idx]);
  else if (idx < 98304)  oC[idx - 65536]  = f2bf(Wc[idx - 65536]);
  else                   oX[idx - 98304]  = f2bf(Wx[idx - 98304]);
}

// M_h[n,e] = sum_d Wk[h*128+d, n] * Wq[h*128+d, e]   (h=2, 256x256 each) -> bf16
__global__ __launch_bounds__(256) void na_prep_m(
    const float* __restrict__ Wq, const float* __restrict__ Wk,
    unsigned short* __restrict__ oM) {
  int idx = blockIdx.x * 256 + threadIdx.x;      // 131072 total
  int h = idx >> 16, n = (idx >> 8) & 255, e = idx & 255;
  const float* kp = Wk + (size_t)(h * 128) * 256 + n;
  const float* qp = Wq + (size_t)(h * 128) * 256 + e;
  float acc = 0.f;
  #pragma unroll 8
  for (int d = 0; d < 128; ++d) acc += kp[d * 256] * qp[d * 256];
  oM[idx] = f2bf(acc);
}

// ---------------- main fused kernel ----------------
#define MB 16        // batch rows per workgroup
#define SN 264       // LDS row stride (bf16 elems): 132 dwords -> conflict-free rows

__global__ __launch_bounds__(256) void na_main(
    const float* __restrict__ center, const float* __restrict__ nbr,
    const float* __restrict__ ew, const float* __restrict__ gamma,
    const float* __restrict__ beta,
    const unsigned short* __restrict__ wM, const unsigned short* __restrict__ wV,
    const unsigned short* __restrict__ wC, const unsigned short* __restrict__ wX,
    float* __restrict__ out) {
  __shared__ unsigned short sNbr[256 * SN];   // 132 KB neighbor tile (bf16)
  __shared__ unsigned short sCen[16 * SN];    // center rows (bf16)
  __shared__ unsigned short sVbar[32 * SN];   // attn-weighted neighbors, per head
  __shared__ unsigned short sCtxN[16 * SN];   // normalized context (bf16)
  __shared__ float sT[2 * 16 * 256];          // t = M_h @ center
  __shared__ float sCtx[16 * 256];            // context (f32, pre-LN)
  __shared__ float sS[512];                   // scores -> attn (in place)
  __shared__ float sEw[256];
  __shared__ float sMu[16], sRi[16];

  const int tid  = threadIdx.x;
  const int lane = tid & 31;
  const int wv   = tid >> 5;                  // wave id 0..7
  const int b0   = blockIdx.x * MB;

  // ---- stage center (16x256) and edge weights ----
  {
    const float* cbase = center + (size_t)b0 * 256;
    for (int i = tid; i < 16 * 256; i += 256)
      sCen[(i >> 8) * SN + (i & 255)] = f2bf(cbase[i]);
    sEw[tid] = ew[(size_t)b0 * 16 + tid];
  }
  // ---- stage neighbors (256x256 f32 -> bf16), non-temporal stream ----
  {
    const f32x4* nb = (const f32x4*)(nbr + (size_t)b0 * 16 * 256);
    for (int i = tid; i < (256 * 256) / 4; i += 256) {
      f32x4 v = __builtin_nontemporal_load(nb + i);
      int flat = i * 4;
      int row = flat >> 8, e = flat & 255;
      ushort4 s;
      s.x = f2bf(v[0]); s.y = f2bf(v[1]); s.z = f2bf(v[2]); s.w = f2bf(v[3]);
      *(ushort4*)(&sNbr[row * SN + e]) = s;
    }
  }
  __syncthreads();

  // ---- t[h][r][n] = sum_e M_h[n,e] * center[r,e] : 32 N-tiles, 4 per wave ----
  for (int j = wv * 4; j < wv * 4 + 4; ++j) {
    int h = j >> 4;
    int n0 = (j & 15) * 16;
    const unsigned short* wMh = wM + (size_t)h * 65536;
    f32x8 acc = {0.f, 0.f, 0.f, 0.f, 0.f, 0.f, 0.f, 0.f};
    #pragma unroll
    for (int k0 = 0; k0 < 256; k0 += 32)
      acc = wmma_bf16(load_a_lds(sCen, SN, k0, lane),
                      load_b_glb(wMh, n0, k0, lane), acc);
    int roff = (lane & 16) ? 8 : 0;
    int n = n0 + (lane & 15);
    #pragma unroll
    for (int g = 0; g < 8; ++g)
      sT[(h * 16 + g + roff) * 256 + n] = acc[g];
  }
  __syncthreads();

  // ---- scores[r][h][k] = scale * t[h][r]·nbr[r,k] + ew[r,k] (VALU) ----
  const float scale = 0.08838834764831845f;   // 128^-0.5
  for (int d = tid; d < 512; d += 256) {
    int r = d >> 5, h = (d >> 4) & 1, k = d & 15;
    const float* tp = &sT[(h * 16 + r) * 256];
    const unsigned short* np = &sNbr[(r * 16 + k) * SN];
    float acc = 0.f;
    #pragma unroll 8
    for (int e = 0; e < 256; ++e) acc += tp[e] * bf2f(np[e]);
    sS[d] = acc * scale + sEw[r * 16 + k];
  }
  __syncthreads();
  // ---- softmax over K=16, one thread per (r,h) ----
  if (tid < 32) {
    int r = tid >> 1, h = tid & 1;
    float* sp = &sS[(r << 5) + (h << 4)];
    float mx = sp[0];
    #pragma unroll
    for (int k = 1; k < 16; ++k) mx = fmaxf(mx, sp[k]);
    float ex[16], sum = 0.f;
    #pragma unroll
    for (int k = 0; k < 16; ++k) { ex[k] = __expf(sp[k] - mx); sum += ex[k]; }
    float inv = 1.f / sum;
    #pragma unroll
    for (int k = 0; k < 16; ++k) sp[k] = ex[k] * inv;
  }
  __syncthreads();

  // ---- vbar[h][r][e] = sum_k attn[r,h,k] * nbr[r,k,e] (VALU) ----
  for (int d = tid; d < 2 * 16 * 256; d += 256) {
    int h = d >> 12, r = (d >> 8) & 15, e = d & 255;
    const float* ap = &sS[(r << 5) + (h << 4)];
    float acc = 0.f;
    #pragma unroll
    for (int k = 0; k < 16; ++k) acc += ap[k] * bf2f(sNbr[(r * 16 + k) * SN + e]);
    sVbar[(h * 16 + r) * SN + e] = f2bf(acc);
  }
  __syncthreads();

  // ---- context[r][h*128+n] = sum_e Wv[h*128+n,e] * vbar[h][r][e] ----
  for (int j = wv * 2; j < wv * 2 + 2; ++j) {
    int h = j >> 3, n0 = (j & 7) * 16;
    f32x8 acc = {0.f, 0.f, 0.f, 0.f, 0.f, 0.f, 0.f, 0.f};
    #pragma unroll
    for (int k0 = 0; k0 < 256; k0 += 32)
      acc = wmma_bf16(load_a_lds(&sVbar[h * 16 * SN], SN, k0, lane),
                      load_b_glb(wV, h * 128 + n0, k0, lane), acc);
    int roff = (lane & 16) ? 8 : 0;
    int col = h * 128 + n0 + (lane & 15);
    #pragma unroll
    for (int g = 0; g < 8; ++g)
      sCtx[(g + roff) * 256 + col] = acc[g];
  }
  __syncthreads();

  // ---- LayerNorm over D=256 ----
  if (tid < 16) {
    const float* cp = &sCtx[tid * 256];
    float s = 0.f, s2 = 0.f;
    for (int e = 0; e < 256; ++e) { float v = cp[e]; s += v; s2 += v * v; }
    float mu = s * (1.f / 256.f);
    float var = s2 * (1.f / 256.f) - mu * mu;
    sMu[tid] = mu;
    sRi[tid] = rsqrtf(var + 1e-5f);
  }
  __syncthreads();
  for (int d = tid; d < 16 * 256; d += 256) {
    int r = d >> 8, e = d & 255;
    float v = (sCtx[d] - sMu[r]) * sRi[r] * gamma[e] + beta[e];
    sCtxN[r * SN + e] = f2bf(v);
  }
  __syncthreads();

  // ---- out = [center @ Wc.T , ctx_n @ Wx.T] ----
  for (int j = wv * 2; j < wv * 2 + 2; ++j) {
    int half = j >> 3, n0 = (j & 7) * 16;
    const unsigned short* A  = half ? sCtxN : sCen;
    const unsigned short* Bw = half ? wX : wC;
    f32x8 acc = {0.f, 0.f, 0.f, 0.f, 0.f, 0.f, 0.f, 0.f};
    #pragma unroll
    for (int k0 = 0; k0 < 256; k0 += 32)
      acc = wmma_bf16(load_a_lds(A, SN, k0, lane),
                      load_b_glb(Bw, n0, k0, lane), acc);
    int roff = (lane & 16) ? 8 : 0;
    int col = half * 128 + n0 + (lane & 15);
    #pragma unroll
    for (int g = 0; g < 8; ++g)
      __builtin_nontemporal_store(acc[g], &out[(size_t)(b0 + g + roff) * 256 + col]);
  }
}

// ---------------- launch ----------------
extern "C" void kernel_launch(void* const* d_in, const int* in_sizes, int n_in,
                              void* d_out, int out_size, void* d_ws, size_t ws_size,
                              hipStream_t stream) {
  const float* center = (const float*)d_in[0];
  const float* nbr    = (const float*)d_in[1];
  const float* ew     = (const float*)d_in[2];
  const float* Wq     = (const float*)d_in[3];
  const float* Wk     = (const float*)d_in[4];
  const float* Wv     = (const float*)d_in[5];
  const float* Wc     = (const float*)d_in[6];
  const float* Wx     = (const float*)d_in[7];
  const float* gamma  = (const float*)d_in[8];
  const float* beta   = (const float*)d_in[9];
  float* out = (float*)d_out;

  unsigned short* ws = (unsigned short*)d_ws;      // 512 KB used
  unsigned short* wM = ws;                         // 2*256*256
  unsigned short* wV = ws + 131072;                // 256*256
  unsigned short* wC = wV + 65536;                 // 128*256
  unsigned short* wX = wC + 32768;                 // 128*256

  na_prep_w<<<512, 256, 0, stream>>>(Wv, Wc, Wx, wV, wC, wX);
  na_prep_m<<<512, 256, 0, stream>>>(Wq, Wk, wM);
  na_main<<<32768 / MB, 256, 0, stream>>>(center, nbr, ew, gamma, beta,
                                          wM, wV, wC, wX, out);
}
